// ConvEventMinGRUCellFuseDownsample_37735582663131
// MI455X (gfx1250) — matrix operand order repaired and checked by reading, not compiled, not run on hardware
//
#include <hip/hip_runtime.h>
#include <hip/hip_bf16.h>

typedef __attribute__((ext_vector_type(16))) _Float16 v16h;
typedef __attribute__((ext_vector_type(8)))  _Float16 v8h;
typedef __attribute__((ext_vector_type(8)))  float    v8f;
typedef __attribute__((ext_vector_type(4)))  int      v4i;

// Address-space-qualified pointer types for the async global->LDS builtin:
// param0 is 'int4 __device__ *' (AS1), param1 is the LDS side (AS3).
typedef __attribute__((address_space(1))) v4i* gptr_v4i;
typedef __attribute__((address_space(3))) v4i* lptr_v4i;

// Problem constants
#define T_STEPS 16
#define NB      8          // batch
#define CH      64         // channels (conv in/out)
#define HW      64         // spatial H == W
#define PAD     66         // padded H/W for staged input
#define S_STEP  ((size_t)NB * CH * HW * HW)   // 2,097,152 elems per (t) per output

// LDS staging: 3 padded rows x 66 pixels x 64 ch, pixel stride padded to 72
// halves (144 B) -> every v16h read stays 16B-aligned and lanes spread banks.
#define LDS_STR 72
#define LDS_ELEMS (3 * PAD * LDS_STR)   // 14,256 halves = 28,512 B per block

// Async global->LDS path (CDNA5 GLOBAL_LOAD_ASYNC_TO_LDS_B128), guarded so the
// file always compiles; fallback is load-to-VGPR + ds_store.
#if defined(__has_builtin)
#  if __has_builtin(__builtin_amdgcn_global_load_async_to_lds_b128)
#    define ASYNC_LDS 1
#  endif
#endif
#ifndef ASYNC_LDS
#  define ASYNC_LDS 0
#endif

// ---------------------------------------------------------------------------
// Weight packing into the 16-bit A-matrix (16x32, MxK) layout used here (contiguous K, matching B)
// lane<16 : M=lane,    halves j: K = j
// lane>=16: M=lane-16, halves j: K = 16+j
// Block per (cot, tap, chunk): 32 lanes x 16 halves (1024 B), lane-contiguous.
// ---------------------------------------------------------------------------
__global__ void pack_weights_kernel(const float* __restrict__ w_rz,
                                    const float* __restrict__ w_f,
                                    _Float16* __restrict__ packRZ,
                                    _Float16* __restrict__ packF) {
  int tid = blockIdx.x * blockDim.x + threadIdx.x;
  if (tid >= 2 * 4 * 9 * 2 * 32 * 16) return;
  int j     = tid & 15;
  int t1    = tid >> 4;
  int lane  = t1 & 31;
  int t2    = t1 >> 5;
  int chunk = t2 & 1;
  int t3    = t2 >> 1;
  int tap   = t3 % 9;
  int t4    = t3 / 9;
  int cot   = t4 & 3;
  int s     = t4 >> 2;

  int M = lane & 15;
  int v = j >> 1, pbit = j & 1;
  int K;
  if (lane < 16) K = 2 * v + pbit;
  else           K = 16 + 2 * v + pbit;

  int co = cot * 16 + M;
  int ci = chunk * 32 + K;
  int ky = tap / 3, kx = tap % 3;

  const float* src = s ? w_f : w_rz;
  _Float16*    dst = s ? packF : packRZ;
  dst[(size_t)((cot * 9 + tap) * 2 + chunk) * 512 + lane * 16 + j] =
      (_Float16)src[((co * CH + ci) * 3 + ky) * 3 + kx];
}

__global__ void thr_kernel(const float* __restrict__ thr_reparam,
                           float* __restrict__ thr) {
  int i = blockIdx.x * blockDim.x + threadIdx.x;
  if (i < HW * HW) thr[i] = 1.0f / (1.0f + __expf(-thr_reparam[i]));
}

// ---------------------------------------------------------------------------
// Cooperative stage of 3 padded input rows (oy..oy+2) into LDS.
// gsrc points at stage + ((n*PAD + oy)*PAD)*CH : 198 pixels x 128 B each.
// ---------------------------------------------------------------------------
__device__ __forceinline__ void stage_to_lds(const _Float16* __restrict__ gsrc,
                                             _Float16* lbuf, int tid) {
  const int NCHUNK = 3 * PAD * 8;  // 1584 x 16B chunks
  for (int c = tid; c < NCHUNK; c += 128) {
    const int p = c >> 3, sub = c & 7;
    const _Float16* g = gsrc + (size_t)p * CH + sub * 8;
    _Float16*       l = lbuf + p * LDS_STR + sub * 8;
#if ASYNC_LDS
    __builtin_amdgcn_global_load_async_to_lds_b128(
        (gptr_v4i)g, (lptr_v4i)l, 0, 0);
#else
    *(v8h*)l = *(const v8h*)g;
#endif
  }
#if ASYNC_LDS
#  if __has_builtin(__builtin_amdgcn_s_wait_asynccnt)
  __builtin_amdgcn_s_wait_asynccnt(0);
#  else
  asm volatile("s_wait_asynccnt 0x0" ::: "memory");
#  endif
#endif
  __syncthreads();
}

// ---------------------------------------------------------------------------
// Conv core: one wave computes co-tile(16) x one output row (4 pixel-tiles of
// 16) as implicit GEMM, K = 9 taps x 64 cin in chunks of 32 via
// v_wmma_f32_16x16x32_f16.  B operands come from the LDS-staged padded-NHWC
// tile; layout: lane n<16 holds column N=n, K=0..15 (contiguous halves);
// lane n+16 holds K=16..31 -> per-lane contiguous 32B (2x ds_load_b128).
// ---------------------------------------------------------------------------
__device__ __forceinline__ void conv_core(const _Float16* lbuf,
                                          const _Float16* __restrict__ packA,
                                          int cot, int lane, v8f acc[4]) {
  const int laneN = lane & 15;
  const int hi    = (lane >> 4) & 1;
  const v16h* __restrict__ A = (const v16h*)packA;
  for (int tap = 0; tap < 9; ++tap) {
    const int dx = tap % 3;
    const int ry = tap / 3;                  // row within 3-row LDS window
    const v16h a0 = A[((cot * 9 + tap) * 2 + 0) * 32 + lane];
    const v16h a1 = A[((cot * 9 + tap) * 2 + 1) * 32 + lane];
    const _Float16* rowp = lbuf + (ry * PAD + dx) * LDS_STR + hi * 16;
#pragma unroll
    for (int pt = 0; pt < 4; ++pt) {
      const _Float16* bp = rowp + (pt * 16 + laneN) * LDS_STR;
      v16h b0 = *(const v16h*)bp;          // cin chunk 0..31 (per-lane half)
      v16h b1 = *(const v16h*)(bp + 32);   // cin chunk 32..63
      acc[pt] = __builtin_amdgcn_wmma_f32_16x16x32_f16(
          false, a0, false, b0, (short)0, acc[pt], false, false);
      acc[pt] = __builtin_amdgcn_wmma_f32_16x16x32_f16(
          false, a1, false, b1, (short)0, acc[pt], false, false);
    }
  }
}

// ---------------------------------------------------------------------------
// Step kernel 1: forget_gate = sigmoid(conv(prev_y, w_rz) + b_rz + xi_i)
// writes Fbuf (f32 NCHW) and stages prev_y*forget (f16 padded NHWC) for conv2.
// ---------------------------------------------------------------------------
__global__ void conv_forget_kernel(const _Float16* __restrict__ stage1,
                                   const _Float16* __restrict__ packRZ,
                                   const float* __restrict__ xt,
                                   const float* __restrict__ b_rz,
                                   const float* __restrict__ prevY,
                                   float* __restrict__ Fbuf,
                                   _Float16* __restrict__ stage2,
                                   int t) {
  __shared__ __attribute__((aligned(32))) _Float16 lbuf[LDS_ELEMS];
  const int n    = blockIdx.x >> 6;
  const int oy   = blockIdx.x & 63;
  const int cot  = threadIdx.x >> 5;
  const int lane = threadIdx.x & 31;

  stage_to_lds(stage1 + ((size_t)n * PAD + oy) * PAD * CH, lbuf, threadIdx.x);

  v8f z8 = {0.f, 0.f, 0.f, 0.f, 0.f, 0.f, 0.f, 0.f};
  v8f acc[4] = {z8, z8, z8, z8};
  conv_core(lbuf, packRZ, cot, lane, acc);

  const int laneN  = lane & 15;
  const int hi     = (lane >> 4) & 1;
  const int cobase = cot * 16 + hi * 8;         // D tile: VGPR r -> M = hi*8+r
  const float* __restrict__ xi =
      xt + ((size_t)(t * NB + n) * (2 * CH)) * (HW * HW) + oy * HW;

  float bias[8];
#pragma unroll
  for (int r = 0; r < 8; ++r) bias[r] = b_rz[cobase + r];

#pragma unroll
  for (int pt = 0; pt < 4; ++pt) {
    const int x = pt * 16 + laneN;              // D tile: N = lane pixel
    v8h pf8;
#pragma unroll
    for (int r = 0; r < 8; ++r) {
      const int co = cobase + r;
      const size_t idx = ((size_t)(n * CH + co)) * (HW * HW) + oy * HW + x;
      float z = acc[pt][r] + bias[r] + xi[(size_t)co * (HW * HW) + x];
      float f = 1.0f / (1.0f + __expf(-z));
      Fbuf[idx] = f;
      pf8[r] = (_Float16)(prevY[idx] * f);
    }
    // NHWC f16: 8 contiguous channels -> one 16B store per lane
    *(v8h*)(stage2 + (((size_t)n * PAD + oy + 1) * PAD + (x + 1)) * CH + cobase) = pf8;
  }
}

// ---------------------------------------------------------------------------
// Step kernel 2: input_gate = tanh(conv(pf, w_f) + b_f + xi_f); GRU update,
// spike threshold, writes ys/events/neg_dists, updates h/prev_y, re-stages y.
// ---------------------------------------------------------------------------
__global__ void conv_update_kernel(const _Float16* __restrict__ stage2,
                                   const _Float16* __restrict__ packF,
                                   const float* __restrict__ xt,
                                   const float* __restrict__ b_f,
                                   const float* __restrict__ Fbuf,
                                   float* __restrict__ Hst,
                                   float* __restrict__ prevY,
                                   const float* __restrict__ thr,
                                   _Float16* __restrict__ stage1,
                                   float* __restrict__ out,
                                   int t) {
  __shared__ __attribute__((aligned(32))) _Float16 lbuf[LDS_ELEMS];
  const int n    = blockIdx.x >> 6;
  const int oy   = blockIdx.x & 63;
  const int cot  = threadIdx.x >> 5;
  const int lane = threadIdx.x & 31;

  stage_to_lds(stage2 + ((size_t)n * PAD + oy) * PAD * CH, lbuf, threadIdx.x);

  v8f z8 = {0.f, 0.f, 0.f, 0.f, 0.f, 0.f, 0.f, 0.f};
  v8f acc[4] = {z8, z8, z8, z8};
  conv_core(lbuf, packF, cot, lane, acc);

  const int laneN  = lane & 15;
  const int hi     = (lane >> 4) & 1;
  const int cobase = cot * 16 + hi * 8;
  const float* __restrict__ xi =
      xt + ((size_t)(t * NB + n) * (2 * CH)) * (HW * HW) + oy * HW;

  const size_t TOT = S_STEP * T_STEPS;
  float* __restrict__ ys = out + (size_t)t * S_STEP;
  float* __restrict__ ev = out + TOT + (size_t)t * S_STEP;
  float* __restrict__ nd = out + 2 * TOT + (size_t)t * S_STEP;

  float bias[8];
#pragma unroll
  for (int r = 0; r < 8; ++r) bias[r] = b_f[cobase + r];

#pragma unroll
  for (int pt = 0; pt < 4; ++pt) {
    const int x  = pt * 16 + laneN;
    const float tv = thr[oy * HW + x];
    v8h y8;
#pragma unroll
    for (int r = 0; r < 8; ++r) {
      const int co = cobase + r;
      const size_t idx = ((size_t)(n * CH + co)) * (HW * HW) + oy * HW + x;
      float z = acc[pt][r] + bias[r] + xi[(size_t)(CH + co) * (HW * HW) + x];
      float g = tanhf(z);
      float f = Fbuf[idx];
      float h = (1.0f - f) * Hst[idx] + f * g;
      float d = h - tv;
      float e = (d > 0.0f) ? 1.0f : 0.0f;
      float y = (d > 0.0f) ? h : 0.0f;
      ys[idx] = y;
      ev[idx] = e;
      nd[idx] = (d < 0.0f) ? (tv - h) : 0.0f;
      Hst[idx]   = h - e * tv;
      prevY[idx] = y;
      y8[r] = (_Float16)y;
    }
    *(v8h*)(stage1 + (((size_t)n * PAD + oy + 1) * PAD + (x + 1)) * CH + cobase) = y8;
  }
}

// ---------------------------------------------------------------------------
extern "C" void kernel_launch(void* const* d_in, const int* in_sizes, int n_in,
                              void* d_out, int out_size, void* d_ws, size_t ws_size,
                              hipStream_t stream) {
  const float* xt    = (const float*)d_in[0];   // [16,8,128,64,64]
  const float* w_rz  = (const float*)d_in[1];   // [64,64,3,3]
  const float* b_rz  = (const float*)d_in[2];   // [64]
  const float* w_f   = (const float*)d_in[3];   // [64,64,3,3]
  const float* b_f   = (const float*)d_in[4];   // [64]
  const float* thr_r = (const float*)d_in[5];   // [64,64]
  (void)in_sizes; (void)n_in; (void)out_size; (void)ws_size;

  const size_t STAGE_BYTES = (size_t)NB * PAD * PAD * CH * 2;  // 4,460,544
  const size_t NCHW_BYTES  = (size_t)NB * CH * HW * HW * 4;    // 8,388,608

  char* p = (char*)d_ws;
  _Float16* stage1 = (_Float16*)(p);
  _Float16* stage2 = (_Float16*)(p + STAGE_BYTES);
  float*    Hst    = (float*)(p + 2 * STAGE_BYTES);
  float*    prevY  = (float*)(p + 2 * STAGE_BYTES + NCHW_BYTES);
  float*    Fbuf   = (float*)(p + 2 * STAGE_BYTES + 2 * NCHW_BYTES);
  float*    thr    = (float*)(p + 2 * STAGE_BYTES + 3 * NCHW_BYTES);
  _Float16* packRZ = (_Float16*)(p + 2 * STAGE_BYTES + 3 * NCHW_BYTES + 16384);
  _Float16* packF  = packRZ + 4 * 9 * 2 * 512;

  // Zero staged inputs (incl. pad borders -> implicit conv zero-padding),
  // h state and prev_y state.  Fbuf is written before read each step.
  (void)hipMemsetAsync(d_ws, 0, 2 * STAGE_BYTES + 2 * NCHW_BYTES, stream);

  thr_kernel<<<(HW * HW + 255) / 256, 256, 0, stream>>>(thr_r, thr);
  pack_weights_kernel<<<(2 * 4 * 9 * 2 * 32 * 16 + 255) / 256, 256, 0, stream>>>(
      w_rz, w_f, packRZ, packF);

  // grid = NB * HW rows; block = 4 waves (one co-tile each), wave32
  for (int t = 0; t < T_STEPS; ++t) {
    conv_forget_kernel<<<NB * HW, 128, 0, stream>>>(
        stage1, packRZ, xt, b_rz, prevY, Fbuf, stage2, t);
    conv_update_kernel<<<NB * HW, 128, 0, stream>>>(
        stage2, packF, xt, b_f, Fbuf, Hst, prevY, thr, stage1, (float*)d_out, t);
  }
}
